// SAGENet_30477087932645
// MI455X (gfx1250) — matrix operand order repaired and checked by reading, not compiled, not run on hardware
//
#include <hip/hip_runtime.h>
#include <hip/hip_bf16.h>

// GraphSAGE (3x SAGEConv+BN+ReLU, residuals) + linear head for gfx1250.
// Strategy:
//   - scatter_kernel: 1 wave per edge, float4 gather + global_atomic_add_f32
//     into agg[] (lives in the 192MB L2), degree count per dst.
//   - gemm_fused: wave-level WMMA bf16 GEMM. Both SAGE GEMMs fused into one
//     K=256 GEMM over [mean | h] with pre-transposed bf16 [Wl;Wr] staged in
//     LDS. Mean division folded into A-load (scale by 1/deg). Epilogue fuses
//     bias + BN(eval) + ReLU + residual. One pass over features per layer.
//   - Software pipelining: all NT B-fragments for a k-chunk are loaded from
//     LDS before the WMMA burst (partial s_wait_dscnt instead of wait-0 per
//     WMMA), and the next k-chunk's A data is fetched during the burst.
//   - __launch_bounds__(256, 1) on the GEMM lifts the VGPR cap so the four
//     A-prefetch b128 loads keep independent destination quads in flight.

typedef __bf16 bf16_t;
typedef __attribute__((ext_vector_type(16))) __bf16 v16bf;
typedef __attribute__((ext_vector_type(8)))  float  v8f;

#define NN     100000
#define DD     128
#define DOUTC  64
#define BN_EPS 1e-5f

// ---------------- edge scatter-add (mean-agg numerator + degree) ----------
__global__ void __launch_bounds__(256)
scatter_kernel(const float* __restrict__ h,
               const long long* __restrict__ src,
               const long long* __restrict__ dst,
               float* __restrict__ agg, float* __restrict__ cnt, int E)
{
    const int wid  = (int)((blockIdx.x * blockDim.x + threadIdx.x) >> 5);
    const int lane = threadIdx.x & 31;
    if (wid >= E) return;
    const int s = (int)src[wid];
    const int d = (int)dst[wid];
    const float4 v = *(const float4*)(h + (size_t)s * DD + lane * 4);
    float* a = agg + (size_t)d * DD + lane * 4;
    atomicAdd(a + 0, v.x);
    atomicAdd(a + 1, v.y);
    atomicAdd(a + 2, v.z);
    atomicAdd(a + 3, v.w);
    if (lane == 0) atomicAdd(cnt + d, 1.0f);
}

// ---------------- 1/max(cnt,1) ------------------------------------------
__global__ void __launch_bounds__(256)
inv_kernel(const float* __restrict__ cnt, float* __restrict__ invc, int n)
{
    const int i = blockIdx.x * blockDim.x + threadIdx.x;
    if (i < n) invc[i] = 1.0f / fmaxf(cnt[i], 1.0f);
}

// ---- pre-transpose weights to [n][k] bf16, K = [Wl rows ; Wr rows] -------
__global__ void __launch_bounds__(256)
prep_w(const float* __restrict__ Wl, const float* __restrict__ Wr,
       bf16_t* __restrict__ out, int K1, int Ktot, int Nc)
{
    const int i = blockIdx.x * blockDim.x + threadIdx.x;
    if (i >= Nc * Ktot) return;
    const int n = i / Ktot, k = i % Ktot;
    const float v = (k < K1) ? Wl[(size_t)k * Nc + n]
                             : Wr[(size_t)(k - K1) * Nc + n];
    out[i] = (bf16_t)v;
}

// ---------------- fused WMMA GEMM + bias/BN/ReLU/residual -----------------
// One wave owns a 16-row x (NT*16)-col output tile; K = KC*32 chunks.
// HAS_AGG: K = [agg*inv | A1] concat (SAGE layers). BN: bias+BN+ReLU+res.
template<int NT, int KC, bool HAS_AGG, bool BN>
__global__ void __launch_bounds__(256, 1)
gemm_fused(const float* __restrict__ A0, const float* __restrict__ invc,
           const float* __restrict__ A1, const bf16_t* __restrict__ Wt,
           const float* __restrict__ bias,
           const float* __restrict__ g,  const float* __restrict__ be,
           const float* __restrict__ rm, const float* __restrict__ rv,
           const float* __restrict__ res, float* __restrict__ out, int M)
{
    constexpr int Nc   = NT * 16;
    constexpr int Ktot = KC * 32;
    __shared__ __align__(32) bf16_t sW[Nc * Ktot];

    {   // cooperative stage of the full weight panel into LDS (<= 64KB)
        const uint4* gs = (const uint4*)Wt;
        uint4*       ls = (uint4*)sW;
        constexpr int nvec = Nc * Ktot * 2 / 16;
        for (int i = threadIdx.x; i < nvec; i += 256) ls[i] = gs[i];
    }
    __syncthreads();

    const int wave = threadIdx.x >> 5;
    const int lane = threadIdx.x & 31;
    const int tile = blockIdx.x * 8 + wave;
    if (tile * 16 >= M) return;              // whole-wave uniform skip
    const int rlo = lane & 15;
    const int hi  = lane >> 4;
    const int row = tile * 16 + rlo;

    v8f zero = {};
    v8f acc[NT];
#pragma unroll
    for (int t = 0; t < NT; ++t) acc[t] = zero;

    const float scale0 = HAS_AGG ? invc[row] : 1.0f;

    // A-chunk fetch: lane<16 holds K {0..15}, lane>=16
    // holds {16..31} within the 32-chunk; one contiguous 16-float run.
    auto loadA = [&](int kc, float4 f[4], float& sc) {
        const float* srcp; int kb;
        if (HAS_AGG && kc >= KC / 2) { srcp = A1; kb = kc * 32 - Ktot / 2; sc = 1.0f;   }
        else                         { srcp = A0; kb = kc * 32;            sc = scale0; }
        const float* p = srcp + (size_t)row * DD + kb + hi * 16;
        f[0] = *(const float4*)(p + 0);
        f[1] = *(const float4*)(p + 4);
        f[2] = *(const float4*)(p + 8);
        f[3] = *(const float4*)(p + 12);
    };

    float4 fcur[4], fnxt[4];
    float  scur, snxt;
    loadA(0, fcur, scur);

#pragma unroll
    for (int kc = 0; kc < KC; ++kc) {
        if (kc + 1 < KC) loadA(kc + 1, fnxt, snxt);   // prefetch next A chunk

        // convert current A chunk to bf16 fragment
        v16bf a;
        const float sc = scur;
#pragma unroll
        for (int q = 0; q < 4; ++q) {
            a[4 * q + 0] = (bf16_t)(fcur[q].x * sc);
            a[4 * q + 1] = (bf16_t)(fcur[q].y * sc);
            a[4 * q + 2] = (bf16_t)(fcur[q].z * sc);
            a[4 * q + 3] = (bf16_t)(fcur[q].w * sc);
        }

        // batch all B fragments first so waitcnt can be partial, then burst WMMAs
        v16bf bfr[NT];
#pragma unroll
        for (int nt = 0; nt < NT; ++nt)
            bfr[nt] = *(const v16bf*)&sW[(size_t)(nt * 16 + rlo) * Ktot + kc * 32 + hi * 16];
#pragma unroll
        for (int nt = 0; nt < NT; ++nt)
            acc[nt] = __builtin_amdgcn_wmma_f32_16x16x32_bf16(
                false, a, false, bfr[nt], (short)0, acc[nt], false, false);

#pragma unroll
        for (int q = 0; q < 4; ++q) fcur[q] = fnxt[q];
        scur = snxt;
    }

    // fused epilogue: bias (+ BN(eval) + ReLU + residual)
#pragma unroll
    for (int nt = 0; nt < NT; ++nt) {
        const int n  = nt * 16 + rlo;
        const float bi = bias[n];
        float sg = 1.0f, sb = 0.0f;
        if (BN) {
            const float istd = rsqrtf(rv[n] + BN_EPS);
            sg = g[n] * istd;
            sb = be[n] - rm[n] * sg;
        }
#pragma unroll
        for (int v = 0; v < 8; ++v) {
            const int m = tile * 16 + v + hi * 8;     // C/D layout: vgpr v, half-wave
            float val = acc[nt][v] + bi;
            if (BN) {
                val = val * sg + sb;
                val = fmaxf(val, 0.0f);
                if (res) val += res[(size_t)m * DD + n];
            }
            out[(size_t)m * Nc + n] = val;
        }
    }
}

extern "C" void kernel_launch(void* const* d_in, const int* in_sizes, int n_in,
                              void* d_out, int out_size, void* d_ws, size_t ws_size,
                              hipStream_t stream)
{
    const float*     x    = (const float*)d_in[0];
    const long long* ei   = (const long long*)d_in[1];   // jnp.int64
    const int        E    = in_sizes[1] / 2;
    const long long* esrc = ei;
    const long long* edst = ei + E;

    const float *Wl[3], *bl[3], *Wr[3], *g[3], *be[3], *rm[3], *rv[3];
    for (int i = 0; i < 3; ++i) {
        const int b = 2 + 7 * i;
        Wl[i] = (const float*)d_in[b + 0];
        bl[i] = (const float*)d_in[b + 1];
        Wr[i] = (const float*)d_in[b + 2];
        g[i]  = (const float*)d_in[b + 3];
        be[i] = (const float*)d_in[b + 4];
        rm[i] = (const float*)d_in[b + 5];
        rv[i] = (const float*)d_in[b + 6];
    }
    const float* Wout = (const float*)d_in[23];
    const float* bout = (const float*)d_in[24];

    // workspace layout
    float*  agg  = (float*)d_ws;                          // N*128
    float*  cnt  = agg  + (size_t)NN * DD;                // N
    float*  invc = cnt  + NN;                             // N
    float*  bufA = invc + NN;                             // N*128
    float*  bufB = bufA + (size_t)NN * DD;                // N*128
    bf16_t* Wt1  = (bf16_t*)(bufB + (size_t)NN * DD);     // 128x256 bf16
    bf16_t* Wt2  = Wt1 + 128 * 256;
    bf16_t* Wt3  = Wt2 + 128 * 256;
    bf16_t* WtO  = Wt3 + 128 * 256;                       // 64x128 bf16

    // weight prep: [n][k] bf16 panels, K = concat(Wl rows, Wr rows)
    prep_w<<<(128 * 256 + 255) / 256, 256, 0, stream>>>(Wl[0], Wr[0], Wt1, 128, 256, 128);
    prep_w<<<(128 * 256 + 255) / 256, 256, 0, stream>>>(Wl[1], Wr[1], Wt2, 128, 256, 128);
    prep_w<<<(128 * 256 + 255) / 256, 256, 0, stream>>>(Wl[2], Wr[2], Wt3, 128, 256, 128);
    prep_w<<<( 64 * 128 + 255) / 256, 256, 0, stream>>>(Wout,  Wout,  WtO, 128, 128,  64);

    const int scatterBlocks = (E * 32 + 255) / 256;   // 1 wave per edge
    const int gemmBlocks    = (NN / 16 + 7) / 8;      // 8 wave-tiles per block

    const float* hin       = x;
    float*       houts[3]  = { bufA, bufB, bufA };
    bf16_t*      Wts[3]    = { Wt1, Wt2, Wt3 };
    const float* resids[3] = { nullptr, bufA, bufB };

    for (int l = 0; l < 3; ++l) {
        hipMemsetAsync(agg, 0, (size_t)NN * DD * sizeof(float), stream);
        hipMemsetAsync(cnt, 0, (size_t)NN * sizeof(float), stream);
        scatter_kernel<<<scatterBlocks, 256, 0, stream>>>(hin, esrc, edst, agg, cnt, E);
        inv_kernel<<<(NN + 255) / 256, 256, 0, stream>>>(cnt, invc, NN);
        gemm_fused<8, 8, true, true><<<gemmBlocks, 256, 0, stream>>>(
            agg, invc, hin, Wts[l], bl[l], g[l], be[l], rm[l], rv[l],
            resids[l], houts[l], NN);
        hin = houts[l];
    }

    // output head: h3 @ Wout + bout  (Nc=64, K=128; no BN/ReLU/residual)
    gemm_fused<4, 4, false, false><<<gemmBlocks, 256, 0, stream>>>(
        bufA, nullptr, nullptr, WtO, bout, nullptr, nullptr, nullptr, nullptr,
        nullptr, (float*)d_out, NN);
}
